// GPTReclinearModel_90074054132188
// MI455X (gfx1250) — matrix operand order repaired and checked by reading, not compiled, not run on hardware
//
#include <hip/hip_runtime.h>
#include <hip/hip_bf16.h>
#include <math.h>

// ---------------------------------------------------------------------------
// Model constants (match reference)
// ---------------------------------------------------------------------------
#define BB   8
#define LL   200
#define DD   512
#define HH   8
#define DH   64
#define DFF  2048
#define NL   4
#define EPS  1e-6f

#define MROWS (BB * LL)          // 1600
#define HELEMS (MROWS * DD)      // 819200

typedef __attribute__((ext_vector_type(16))) __bf16 v16bf;
typedef __attribute__((ext_vector_type(8)))  float  v8f;

// ---------------------------------------------------------------------------
// Embedding gather: h = item_emb[inputs] + user_emb[user_id+1] + pos + seg
// ---------------------------------------------------------------------------
__global__ __launch_bounds__(256)
void embed_kernel(const float* __restrict__ item_emb,
                  const float* __restrict__ user_emb,
                  const float* __restrict__ pos_emb,
                  const float* __restrict__ seg_emb,
                  const int*   __restrict__ user_id,
                  const int*   __restrict__ inputs,
                  const int*   __restrict__ segments,
                  float* __restrict__ hout)
{
    int idx = blockIdx.x * blockDim.x + threadIdx.x;
    if (idx >= HELEMS) return;
    int d  = idx & (DD - 1);
    int bl = idx / DD;
    int l  = bl % LL;
    int b  = bl / LL;
    int it  = inputs[b * LL + l];
    int pos = (it == 0) ? 0 : l;
    int seg = segments[b * LL + l];
    int u   = user_id[b] + 1;
    hout[idx] = item_emb[(size_t)it * DD + d]
              + user_emb[(size_t)u  * DD + d]
              + pos_emb[(size_t)pos * DD + d]
              + seg_emb[(size_t)seg * DD + d];
}

// ---------------------------------------------------------------------------
// WMMA GEMM:  C[M,N] = act( A[M,K] * W[N,K]^T + bias[N] )
//   A, W fp32 in memory, converted inline to bf16 fragments.
//   act: 0 = identity, 1 = sigmoid, 2 = exact GELU
//
// Block = 128 threads (4 waves). Each wave: 16 rows x 64 cols (4 wmma tiles),
// block tile = 64x64, K-loop step 32 with one A fragment reused by 4 WMMAs.
//
// Ragged N edge (N % 64 != 0, N % 16 == 0): the K-loop is branch-free —
// invalid 16-wide tiles have their W pointer clamped to the last valid tile
// (valid memory, dead result) and only the *stores* are predicated off with a
// wave-uniform condition, so EXEC is all-ones for every WMMA as required.
// ---------------------------------------------------------------------------
__global__ __launch_bounds__(128)
void gemm_bf16_wmma(const float* __restrict__ A,
                    const float* __restrict__ W,
                    const float* __restrict__ bias,
                    float* __restrict__ C,
                    int M, int N, int K, int act)
{
    const int lane = threadIdx.x & 31;
    const int wave = threadIdx.x >> 5;

    const int m0    = blockIdx.y * 64 + wave * 16;   // wave's row strip
    const int n_blk = blockIdx.x * 64;

    const int half = lane >> 4;          // 0 or 1
    const int lid  = lane & 15;

    // A fragment: row = m0 + lid ; elems 0..15 -> k0 + half*16 + i
    const float* Arow = A + (size_t)(m0 + lid) * K + half * 16;
    // B fragment: col = n0 + lid ; elems 0..15 -> k0 + half*16 + i
    const int kB = half * 16;

    v8f acc[4] = {};

    const float* Wrow[4];
    bool store_ok[4];
#pragma unroll
    for (int j = 0; j < 4; ++j) {
        int n0 = n_blk + j * 16;
        store_ok[j] = (n0 < N);                    // N % 16 == 0 always
        int n0c = store_ok[j] ? n0 : (N - 16);     // clamp loads to valid rows
        Wrow[j] = W + (size_t)(n0c + lid) * K + kB;
    }

    for (int k0 = 0; k0 < K; k0 += 32) {
        v16bf afrag;
        {
            const float* p0 = Arow + k0;
            const float* p1 = Arow + k0 + 8;
#pragma unroll
            for (int i = 0; i < 8; ++i) afrag[i]     = (__bf16)p0[i];
#pragma unroll
            for (int i = 0; i < 8; ++i) afrag[8 + i] = (__bf16)p1[i];
        }
#pragma unroll
        for (int j = 0; j < 4; ++j) {
            const float* p = Wrow[j] + k0;
            v16bf bfrag;
#pragma unroll
            for (int i = 0; i < 16; ++i) bfrag[i] = (__bf16)p[i];
            acc[j] = __builtin_amdgcn_wmma_f32_16x16x32_bf16(
                         false, afrag, false, bfrag,
                         (short)0, acc[j], false, false);
        }
    }

    // D layout: col = n0 + lid ; row = m0 + half*8 + i
    const int mrow = m0 + half * 8;
#pragma unroll
    for (int j = 0; j < 4; ++j) {
        if (!store_ok[j]) continue;                // wave-uniform
        int n = n_blk + j * 16 + lid;
        float bv = bias ? bias[n] : 0.0f;
#pragma unroll
        for (int i = 0; i < 8; ++i) {
            float v = acc[j][i] + bv;
            if (act == 1)      v = 1.0f / (1.0f + __expf(-v));
            else if (act == 2) v = 0.5f * v * (1.0f + erff(v * 0.70710678118654752f));
            C[(size_t)(mrow + i) * N + n] = v;
        }
    }
}

// ---------------------------------------------------------------------------
// Flowformer linear-attention scan. One 64-thread block per (b,h).
// Thread t owns: scan component d=t of the cumulative vectors, and kv column
// m=t (64 f32 registers). Sequential over l (true data dependence).
// q,k already sigmoid'ed by the projection GEMMs.
// ---------------------------------------------------------------------------
__global__ __launch_bounds__(64)
void flow_attn(const float* __restrict__ Q,
               const float* __restrict__ Km,
               const float* __restrict__ V,
               float* __restrict__ O)
{
    const int bh = blockIdx.x;
    const int b  = bh / HH;
    const int h  = bh % HH;
    const int t  = threadIdx.x;

    __shared__ float red1[64];
    __shared__ float red2[64];
    __shared__ float ksh[64];
    __shared__ float qnsh[64];

    float kv[DH];
#pragma unroll
    for (int d = 0; d < DH; ++d) kv[d] = 0.0f;

    float cumk = 0.f, cumq = 0.f, cumks = 0.f, cumqs = 0.f, cume = 0.f;

    const size_t base = ((size_t)b * LL) * DD + (size_t)h * DH + t;

    for (int l = 0; l < LL; ++l) {
        const size_t idx = base + (size_t)l * DD;
        float q  = Q[idx];
        float kk = Km[idx];
        float v  = V[idx];

        cumk += kk; cumq += q;
        red1[t] = (q  + EPS) * (cumk + EPS);
        red2[t] = (kk + EPS) * (cumq + EPS);
        __syncthreads();
        float S1 = 0.f, S2 = 0.f;
        for (int d = 0; d < 64; ++d) { S1 += red1[d]; S2 += red2[d]; }
        __syncthreads();

        const float n = (float)(l + 1);
        const float sink = n / S1;
        const float src  = n / S2;

        cumks += kk * src;
        cumqs += q  * sink;
        red1[t] = (q  + EPS) * (cumks + EPS);
        red2[t] = (kk + EPS) * (cumqs + EPS);
        __syncthreads();
        float S3 = 0.f, S4 = 0.f;
        for (int d = 0; d < 64; ++d) { S3 += red1[d]; S4 += red2[d]; }
        __syncthreads();

        const float c_sink = S3 / n;
        const float c_src  = fminf(fmaxf(S4 / n, -1.0f), 1.0f);
        const float alloc  = 1.0f / (1.0f + __expf(-c_sink));
        const float e      = __expf(c_src);
        cume += e;
        const float comp = e / cume * n;

        const float qn = q * (sink / n);
        const float vn = v * comp;

        ksh[t]  = kk;
        qnsh[t] = qn;
        __syncthreads();

        float o = 0.f;
#pragma unroll 8
        for (int d = 0; d < DH; ++d) {
            kv[d] += ksh[d] * vn;          // kv[d][m=t] cumsum
            o     += qnsh[d] * kv[d];      // o[l][m=t]
        }
        O[idx] = o * alloc;
        __syncthreads();
    }
}

// ---------------------------------------------------------------------------
// Fused residual add + LayerNorm over D=512. One block per token row.
// Y = (x - mean)/sqrt(var + 1e-5) * g + b,  x = X + R.   (Y may alias X.)
// ---------------------------------------------------------------------------
__global__ __launch_bounds__(128)
void add_layernorm(const float* __restrict__ X,
                   const float* __restrict__ R,
                   const float* __restrict__ g,
                   const float* __restrict__ bta,
                   float* __restrict__ Y)
{
    const int row = blockIdx.x;
    const int tid = threadIdx.x;
    const float* x = X + (size_t)row * DD;
    const float* r = R + (size_t)row * DD;
    float*       y = Y + (size_t)row * DD;

    __shared__ float sbuf[128];

    float v[4];
    float s = 0.f;
#pragma unroll
    for (int i = 0; i < 4; ++i) {
        int d = tid + i * 128;
        v[i] = x[d] + r[d];
        s += v[i];
    }
    sbuf[tid] = s;
    __syncthreads();
    for (int off = 64; off > 0; off >>= 1) {
        if (tid < off) sbuf[tid] += sbuf[tid + off];
        __syncthreads();
    }
    const float mean = sbuf[0] * (1.0f / DD);
    __syncthreads();

    float s2 = 0.f;
#pragma unroll
    for (int i = 0; i < 4; ++i) {
        float dv = v[i] - mean;
        s2 += dv * dv;
    }
    sbuf[tid] = s2;
    __syncthreads();
    for (int off = 64; off > 0; off >>= 1) {
        if (tid < off) sbuf[tid] += sbuf[tid + off];
        __syncthreads();
    }
    const float rstd = rsqrtf(sbuf[0] * (1.0f / DD) + 1e-5f);

#pragma unroll
    for (int i = 0; i < 4; ++i) {
        int d = tid + i * 128;
        y[d] = (v[i] - mean) * rstd * g[d] + bta[d];
    }
}

// ---------------------------------------------------------------------------
// Plain f32 copy (h -> first output region)
// ---------------------------------------------------------------------------
__global__ __launch_bounds__(256)
void copy_f32(const float* __restrict__ src, float* __restrict__ dst, int n)
{
    int i = blockIdx.x * blockDim.x + threadIdx.x;
    if (i < n) dst[i] = src[i];
}

// ---------------------------------------------------------------------------
// Launcher
// ---------------------------------------------------------------------------
extern "C" void kernel_launch(void* const* d_in, const int* in_sizes, int n_in,
                              void* d_out, int out_size, void* d_ws, size_t ws_size,
                              hipStream_t stream)
{
    (void)in_sizes; (void)n_in; (void)out_size; (void)ws_size;

    const float* item_emb = (const float*)d_in[0];
    const float* user_emb = (const float*)d_in[1];
    const float* pos_emb  = (const float*)d_in[2];
    const float* seg_emb  = (const float*)d_in[3];
    const float* aWq = (const float*)d_in[4];
    const float* abq = (const float*)d_in[5];
    const float* aWk = (const float*)d_in[6];
    const float* abk = (const float*)d_in[7];
    const float* aWv = (const float*)d_in[8];
    const float* abv = (const float*)d_in[9];
    const float* aWo = (const float*)d_in[10];
    const float* abo = (const float*)d_in[11];
    const float* ln1g = (const float*)d_in[12];
    const float* ln1b = (const float*)d_in[13];
    const float* fW1  = (const float*)d_in[14];
    const float* fb1  = (const float*)d_in[15];
    const float* fW2  = (const float*)d_in[16];
    const float* fb2  = (const float*)d_in[17];
    const float* ln2g = (const float*)d_in[18];
    const float* ln2b = (const float*)d_in[19];
    const float* outW = (const float*)d_in[20];
    const float* outb = (const float*)d_in[21];
    const int* user_id  = (const int*)d_in[22];
    const int* inputs   = (const int*)d_in[23];
    const int* segments = (const int*)d_in[24];

    // Workspace layout (floats)
    float* ws   = (float*)d_ws;
    float* h    = ws;                         // 819200
    float* qb   = h   + HELEMS;               // 819200
    float* kb   = qb  + HELEMS;               // 819200
    float* vb   = kb  + HELEMS;               // 819200
    float* ob   = vb  + HELEMS;               // 819200
    float* tmp  = ob  + HELEMS;               // 819200
    float* mid  = tmp + HELEMS;               // 1600*2048

    float* out_h      = (float*)d_out;
    float* out_logits = out_h + HELEMS;

    // 1) embeddings
    embed_kernel<<<(HELEMS + 255) / 256, 256, 0, stream>>>(
        item_emb, user_emb, pos_emb, seg_emb, user_id, inputs, segments, h);

    const dim3 blk(128);
    const dim3 grid_d (DD   / 64, MROWS / 64);   // N=512  -> (8, 25)
    const dim3 grid_ff(DFF  / 64, MROWS / 64);   // N=2048 -> (32, 25)

    for (int i = 0; i < NL; ++i) {
        const float* Wq = aWq + (size_t)i * DD * DD;
        const float* Wk = aWk + (size_t)i * DD * DD;
        const float* Wv = aWv + (size_t)i * DD * DD;
        const float* Wo = aWo + (size_t)i * DD * DD;

        // q = sigmoid(h Wq^T + bq), k = sigmoid(...), v = h Wv^T + bv
        gemm_bf16_wmma<<<grid_d, blk, 0, stream>>>(h, Wq, abq + i * DD, qb,
                                                   MROWS, DD, DD, 1);
        gemm_bf16_wmma<<<grid_d, blk, 0, stream>>>(h, Wk, abk + i * DD, kb,
                                                   MROWS, DD, DD, 1);
        gemm_bf16_wmma<<<grid_d, blk, 0, stream>>>(h, Wv, abv + i * DD, vb,
                                                   MROWS, DD, DD, 0);

        // Flowformer causal linear attention scan
        flow_attn<<<BB * HH, 64, 0, stream>>>(qb, kb, vb, ob);

        // output projection
        gemm_bf16_wmma<<<grid_d, blk, 0, stream>>>(ob, Wo, abo + i * DD, tmp,
                                                   MROWS, DD, DD, 0);
        // h = LN(h + attn)
        add_layernorm<<<MROWS, 128, 0, stream>>>(h, tmp, ln1g + i * DD,
                                                 ln1b + i * DD, h);

        // FFN: mid = gelu(h W1^T + b1) ; tmp = mid W2^T + b2
        gemm_bf16_wmma<<<grid_ff, blk, 0, stream>>>(h, fW1 + (size_t)i * DFF * DD,
                                                    fb1 + i * DFF, mid,
                                                    MROWS, DFF, DD, 2);
        gemm_bf16_wmma<<<grid_d, blk, 0, stream>>>(mid, fW2 + (size_t)i * DD * DFF,
                                                   fb2 + i * DD, tmp,
                                                   MROWS, DD, DFF, 0);
        // h = LN(h + ffn)
        add_layernorm<<<MROWS, 128, 0, stream>>>(h, tmp, ln2g + i * DD,
                                                 ln2b + i * DD, h);
    }

    // outputs: (h, logits)
    copy_f32<<<(HELEMS + 255) / 256, 256, 0, stream>>>(h, out_h, HELEMS);

    const int NV = 30000;
    const dim3 grid_v((NV + 63) / 64, MROWS / 64);   // (469, 25)
    gemm_bf16_wmma<<<grid_v, blk, 0, stream>>>(h, outW, outb, out_logits,
                                               MROWS, NV, DD, 0);
}
